// GhostLinear_85392539779590
// MI455X (gfx1250) — compile-verified, matrix operand order checked by reading, not run on hardware
//
#include <hip/hip_runtime.h>

// GhostLinear: out[M,N] = x[M,K] @ (LUT[idx]*scale)[N,K]^T
//   M = B*S = 4096, N = 8192, K = 2048 (fp32 reference).
// Strategy: dequantize weights + convert x to bf16 in d_ws (needs 48 MB),
// then bf16 WMMA GEMM, fp32 accum, with CDNA5 async global->LDS staging.

#define M_DIM 4096
#define N_DIM 8192
#define K_DIM 2048

#define BM 128
#define BN 128
#define BK 32
#define LDS_STRIDE 40              // BK + 8 ushort pad: no bank conflicts, 16B aligned rows
#define BUF_ELTS (BM * LDS_STRIDE) // one buffer, in ushorts
#define NSTEPS (K_DIM / BK)

typedef __attribute__((ext_vector_type(16))) __bf16       v16bf;
typedef __attribute__((ext_vector_type(8)))  float        v8f;
typedef __attribute__((ext_vector_type(4)))  unsigned int u32x4;
typedef __attribute__((ext_vector_type(4)))  float        f32x4;
typedef __attribute__((ext_vector_type(4)))  int          i32x4;

// types for the async-to-LDS builtin: param0 is 'v4i __device__*' (AS1),
// param1 is the LDS-side pointer (AS3)
typedef int v4i __attribute__((vector_size(16)));
typedef __attribute__((address_space(1))) v4i as1_v4i;
typedef __attribute__((address_space(3))) v4i as3_v4i;

__constant__ float c_lut[15] = {
    -1.0f, -0.5f, -0.333333f, -0.2f, -0.142857f, -0.090909f, -0.076923f,
     0.0f,  0.076923f, 0.090909f, 0.142857f, 0.2f, 0.333333f, 0.5f, 1.0f};

__device__ __forceinline__ unsigned int f2bf(float f) {
  unsigned int u = __builtin_bit_cast(unsigned int, f);
  u += 0x7FFFu + ((u >> 16) & 1u);   // round-to-nearest-even
  return u >> 16;
}

// ---- Pre-pass 1: x fp32 -> bf16 ------------------------------------------
__global__ void __launch_bounds__(256) cvt_x_bf16(
    const float* __restrict__ x, unsigned short* __restrict__ xb) {
  const int i = (blockIdx.x * 256 + threadIdx.x) * 8;
  f32x4 a = *reinterpret_cast<const f32x4*>(x + i);
  f32x4 b = *reinterpret_cast<const f32x4*>(x + i + 4);
  u32x4 o;
  o.x = f2bf(a.x) | (f2bf(a.y) << 16);
  o.y = f2bf(a.z) | (f2bf(a.w) << 16);
  o.z = f2bf(b.x) | (f2bf(b.y) << 16);
  o.w = f2bf(b.z) | (f2bf(b.w) << 16);
  *reinterpret_cast<u32x4*>(xb + i) = o;
}

// ---- Pre-pass 2: int32 LUT index -> bf16 weight --------------------------
__global__ void __launch_bounds__(256) deq_w_bf16(
    const int* __restrict__ idx, const float* __restrict__ scale,
    unsigned short* __restrict__ wb) {
  const float s = *scale;
  const int i = (blockIdx.x * 256 + threadIdx.x) * 8;
  i32x4 a = *reinterpret_cast<const i32x4*>(idx + i);
  i32x4 b = *reinterpret_cast<const i32x4*>(idx + i + 4);
  u32x4 o;
  o.x = f2bf(c_lut[a.x] * s) | (f2bf(c_lut[a.y] * s) << 16);
  o.y = f2bf(c_lut[a.z] * s) | (f2bf(c_lut[a.w] * s) << 16);
  o.z = f2bf(c_lut[b.x] * s) | (f2bf(c_lut[b.y] * s) << 16);
  o.w = f2bf(c_lut[b.z] * s) | (f2bf(c_lut[b.w] * s) << 16);
  *reinterpret_cast<u32x4*>(wb + i) = o;
}

// ---- CDNA5 async global->LDS staging -------------------------------------
#if defined(__gfx1250__) && __has_builtin(__builtin_amdgcn_global_load_async_to_lds_b128)
#define HAVE_ASYNC_LDS 1
#else
#define HAVE_ASYNC_LDS 0
#endif

__device__ __forceinline__ as1_v4i* glb_v4i_ptr(const unsigned short* p) {
  // generic global address == AS1 address
  return (as1_v4i*)(uintptr_t)p;
}
__device__ __forceinline__ as3_v4i* lds_v4i_ptr(unsigned short* p) {
  // low 32 bits of a generic LDS pointer are the LDS byte address (AS3)
  return (as3_v4i*)(unsigned int)(uintptr_t)p;
}

__device__ __forceinline__ void cp_b128(const unsigned short* g, unsigned short* l) {
#if HAVE_ASYNC_LDS
  __builtin_amdgcn_global_load_async_to_lds_b128(glb_v4i_ptr(g), lds_v4i_ptr(l), 0, 0);
#else
  *reinterpret_cast<u32x4*>(l) = *reinterpret_cast<const u32x4*>(g);
#endif
}

__device__ __forceinline__ void wait_async0() {
#if HAVE_ASYNC_LDS
#if __has_builtin(__builtin_amdgcn_s_wait_asynccnt)
  __builtin_amdgcn_s_wait_asynccnt(0);
#else
  asm volatile("s_wait_asynccnt 0" ::: "memory");
#endif
#endif
}

// ---- Main GEMM: C[M,N] = A[M,K] * B[N,K]^T, bf16 in / f32 out ------------
union FragU { u32x4 q[2]; v16bf v; };   // two ds_load_b128 -> one 16x32 bf16 fragment

__global__ void __launch_bounds__(256) gemm_wmma_bf16(
    const unsigned short* __restrict__ A,   // [M,K] bf16
    const unsigned short* __restrict__ B,   // [N,K] bf16
    float* __restrict__ C) {                // [M,N] f32
  __shared__ __align__(16) unsigned short As[2 * BUF_ELTS];
  __shared__ __align__(16) unsigned short Bs[2 * BUF_ELTS];

  const int tid   = threadIdx.x;
  const int wave  = tid >> 5;          // 8 waves
  const int lane  = tid & 31;
  const int wm    = wave & 1;          // 2 wave-rows (64 M each)
  const int wn    = wave >> 1;         // 4 wave-cols (32 N each)
  const int lmod  = lane & 15;         // fragment row/col within 16
  const int lhalf = (lane >> 4) * 8;   // WMMA K sub-offset: 0 or 8

  const int mBase = blockIdx.y * BM;
  const int nBase = blockIdx.x * BN;

  // staging map: thread covers rows r and r+64, one 8-elem K chunk
  const int r  = tid >> 2;
  const int kc = (tid & 3) * 8;
  const unsigned short* gA = A + (mBase + r) * K_DIM + kc;
  const unsigned short* gB = B + (nBase + r) * K_DIM + kc;
  const int s0 = r * LDS_STRIDE + kc;
  const int s1 = s0 + 64 * LDS_STRIDE;

  v8f acc[4][2] = {};

  // stage tile 0 into buffer 0
  cp_b128(gA,              &As[s0]);
  cp_b128(gA + 64 * K_DIM, &As[s1]);
  cp_b128(gB,              &Bs[s0]);
  cp_b128(gB + 64 * K_DIM, &Bs[s1]);
  wait_async0();
  __syncthreads();

  int cur = 0;
#pragma unroll 1
  for (int kk = 0; kk < NSTEPS; ++kk) {
    const int nxt  = cur ^ BUF_ELTS;
    const bool more = (kk + 1) < NSTEPS;

    if (more) {
      const unsigned short* nA = gA + (kk + 1) * BK;
      const unsigned short* nB = gB + (kk + 1) * BK;
      cp_b128(nA,              &As[nxt + s0]);
      cp_b128(nA + 64 * K_DIM, &As[nxt + s1]);
      cp_b128(nB,              &Bs[nxt + s0]);
      cp_b128(nB + 64 * K_DIM, &Bs[nxt + s1]);
      if (kk + 2 < NSTEPS) {      // warm L2 two tiles ahead (global_prefetch_b8)
        __builtin_prefetch(gA + (kk + 2) * BK, 0, 3);
        __builtin_prefetch(gB + (kk + 2) * BK, 0, 3);
      }
    }

    const unsigned short* as = &As[cur];
    const unsigned short* bs = &Bs[cur];

    // Load ALL fragments first (distinct registers) so the 12 ds_load_b128
    // pipeline instead of stalling on a reused register set, then drain
    // the 8 WMMAs.
    FragU bfr[2], af[4];
#pragma unroll
    for (int j = 0; j < 2; ++j) {
      const int off = (wn * 32 + j * 16 + lmod) * LDS_STRIDE + lhalf;
      bfr[j].q[0] = *reinterpret_cast<const u32x4*>(bs + off);       // K 0..7 / 8..15
      bfr[j].q[1] = *reinterpret_cast<const u32x4*>(bs + off + 16);  // K 16..23 / 24..31
    }
#pragma unroll
    for (int i = 0; i < 4; ++i) {
      const int off = (wm * 64 + i * 16 + lmod) * LDS_STRIDE + lhalf;
      af[i].q[0] = *reinterpret_cast<const u32x4*>(as + off);
      af[i].q[1] = *reinterpret_cast<const u32x4*>(as + off + 16);
    }
#pragma unroll
    for (int i = 0; i < 4; ++i) {
      acc[i][0] = __builtin_amdgcn_wmma_f32_16x16x32_bf16(
          false, af[i].v, false, bfr[0].v, (short)0, acc[i][0], false, false);
      acc[i][1] = __builtin_amdgcn_wmma_f32_16x16x32_bf16(
          false, af[i].v, false, bfr[1].v, (short)0, acc[i][1], false, false);
    }

    if (more) {
      wait_async0();      // next tile fully landed in LDS
      __syncthreads();    // everyone done reading cur / next visible
    }
    cur = nxt;
  }

  // epilogue: C/D layout — lane holds col lmod, rows v + lhalf
#pragma unroll
  for (int i = 0; i < 4; ++i) {
#pragma unroll
    for (int j = 0; j < 2; ++j) {
      const int col = nBase + wn * 32 + j * 16 + lmod;
      float* cp = C + (size_t)(mBase + wm * 64 + i * 16 + lhalf) * N_DIM + col;
#pragma unroll
      for (int v = 0; v < 8; ++v) {
        cp[(size_t)v * N_DIM] = acc[i][j][v];
      }
    }
  }
}

extern "C" void kernel_launch(void* const* d_in, const int* in_sizes, int n_in,
                              void* d_out, int out_size, void* d_ws, size_t ws_size,
                              hipStream_t stream) {
  const float* x     = (const float*)d_in[0];   // [2,2048,2048] fp32
  const int*   gidx  = (const int*)d_in[1];     // [8192,2048] int32 in [0,15)
  const float* scale = (const float*)d_in[2];   // scalar
  float* out = (float*)d_out;                   // [2,2048,8192] fp32

  // workspace: xb (M*K bf16 = 16 MB) | wb (N*K bf16 = 32 MB)
  unsigned short* xb = (unsigned short*)d_ws;
  unsigned short* wb = xb + (size_t)M_DIM * K_DIM;

  cvt_x_bf16<<<(M_DIM * K_DIM) / (256 * 8), 256, 0, stream>>>(x, xb);
  deq_w_bf16<<<(N_DIM * K_DIM) / (256 * 8), 256, 0, stream>>>(gidx, scale, wb);

  dim3 grid(N_DIM / BN, M_DIM / BM);
  gemm_wmma_bf16<<<grid, 256, 0, stream>>>(xb, wb, out);
}
